// Gradformer_62173946576948
// MI455X (gfx1250) — compile-verified, matrix operand order checked by reading, not run on hardware
//
#include <hip/hip_runtime.h>

#define HID 128
#define NNODES 16384
#define NEDGES 262144
#define NB 128
#define SS 128
#define NHEADS 4
#define DKH 32

typedef __attribute__((ext_vector_type(16))) __bf16 v16bf;
typedef __attribute__((ext_vector_type(8)))  __bf16 v8bf;
typedef __attribute__((ext_vector_type(8)))  float  v8f;
typedef __attribute__((ext_vector_type(4)))  float  v4f;

union BFV { v16bf v; v8bf h[2]; };

#define WMMA_BF16(A, B, C) \
  __builtin_amdgcn_wmma_f32_16x16x32_bf16(false, (A), false, (B), (short)0, (C), false, false)

// ---------------------------------------------------------------------------
// Generic C[M,128] = A[M,128] @ W[128,128] + bias   (WMMA bf16, f32 accum)
// block: 256 threads = 8 waves; each wave: 16 rows x 128 cols; grid: M/128
// ---------------------------------------------------------------------------
__global__ void k_gemm128(const float* __restrict__ A,
                          const float* __restrict__ W,
                          const float* __restrict__ bias,
                          float* __restrict__ C, int M)
{
  __shared__ __bf16 Wt[HID * HID];           // transposed: Wt[n*128 + k]
  const int tid = threadIdx.x;
  for (int i = tid; i < HID * HID; i += 256) {
    int k = i >> 7, n = i & 127;
    Wt[n * HID + k] = (__bf16)W[i];
  }
  __syncthreads();

  const int wave = tid >> 5;
  const int lane = tid & 31;
  const int lrow = lane & 15;
  const int half = lane >> 4;
  const long rowBase = (long)blockIdx.x * 128 + wave * 16;

  v8f acc[8] = {};
  const float* arow = A + (rowBase + lrow) * HID;

  for (int kt = 0; kt < 4; ++kt) {
    const int kb = kt * 32 + half * 8;
    v4f a0 = *(const v4f*)(arow + kb);
    v4f a1 = *(const v4f*)(arow + kb + 4);
    v4f a2 = *(const v4f*)(arow + kb + 16);
    v4f a3 = *(const v4f*)(arow + kb + 20);
    v16bf af;
#pragma unroll
    for (int j = 0; j < 4; ++j) {
      af[j]      = (__bf16)a0[j];
      af[4 + j]  = (__bf16)a1[j];
      af[8 + j]  = (__bf16)a2[j];
      af[12 + j] = (__bf16)a3[j];
    }
#pragma unroll
    for (int nt = 0; nt < 8; ++nt) {
      const __bf16* wcol = &Wt[(nt * 16 + lrow) * HID + kb];
      BFV bfr;
      bfr.h[0] = *(const v8bf*)(wcol);
      bfr.h[1] = *(const v8bf*)(wcol + 16);
      acc[nt] = WMMA_BF16(af, bfr.v, acc[nt]);
    }
  }
#pragma unroll
  for (int nt = 0; nt < 8; ++nt) {
    const int col = nt * 16 + lrow;
    const float bv = bias[col];
#pragma unroll
    for (int r = 0; r < 8; ++r) {
      long m = rowBase + r + half * 8;
      C[m * HID + col] = acc[nt][r] + bv;
    }
  }
}

// ---------------------------------------------------------------------------
// h = concat(x @ node_w + node_b, pe @ pe_w + pe_b)   grid: N blocks x 128 thr
// ---------------------------------------------------------------------------
__global__ void k_embed(const float* __restrict__ x, const float* __restrict__ pe,
                        const float* __restrict__ nw, const float* __restrict__ nb,
                        const float* __restrict__ pw, const float* __restrict__ pb,
                        float* __restrict__ h)
{
  const int n = blockIdx.x, c = threadIdx.x;
  float acc;
  if (c < 112) {
    acc = nb[c];
    const float* xr = x + (long)n * 64;
    for (int j = 0; j < 64; ++j) acc += xr[j] * nw[j * 112 + c];
  } else {
    const int c2 = c - 112;
    acc = pb[c2];
    const float* pr = pe + (long)n * 16;
    for (int j = 0; j < 16; ++j) acc += pr[j] * pw[j * 16 + c2];
  }
  h[(long)n * HID + c] = acc;
}

__global__ void k_copy(const float* __restrict__ a, float* __restrict__ b, long long n)
{
  long long i = (long long)blockIdx.x * 256 + threadIdx.x;
  if (i < n) b[i] = a[i];
}

__global__ void k_zero(float* __restrict__ p, long long n)
{
  long long i = (long long)blockIdx.x * 256 + threadIdx.x;
  if (i < n) p[i] = 0.f;
}

// ---------------------------------------------------------------------------
// e_ij = Dx[dst] + Ex[src] + Ce (in place in eij); sigma=sigmoid;
// atomic scatter num += sigma*Bx[src], den += sigma. One thread per (edge,ch).
// ---------------------------------------------------------------------------
__global__ void k_edge_gate(const float* __restrict__ Bx, const float* __restrict__ Dx,
                            const float* __restrict__ Ex, float* __restrict__ eij,
                            const int* __restrict__ src, const int* __restrict__ dst,
                            float* __restrict__ num, float* __restrict__ den)
{
  long long idx = (long long)blockIdx.x * 256 + threadIdx.x;
  const long long eidx = idx >> 7;
  const int c = (int)(idx & 127);
  const int s = src[eidx];
  const int d = dst[eidx];
  float val = eij[idx] + Dx[(long)d * HID + c] + Ex[(long)s * HID + c];
  eij[idx] = val;
  const float sig = 1.f / (1.f + __expf(-val));
  atomicAdd(&num[(long)d * HID + c], sig * Bx[(long)s * HID + c]);
  atomicAdd(&den[(long)d * HID + c], sig);
}

__global__ void k_hnew(const float* __restrict__ Ax, const float* __restrict__ num,
                       const float* __restrict__ den, float* __restrict__ hnew)
{
  long long i = (long long)blockIdx.x * 256 + threadIdx.x;
  hnew[i] = Ax[i] + num[i] / (den[i] + 1e-6f);
}

// per-channel sum & sumsq (atomics); grid: M/128 blocks x 128 threads
__global__ void k_bn_stats(const float* __restrict__ buf, float* __restrict__ stats)
{
  const int c = threadIdx.x;
  const long base = (long)blockIdx.x * 128;
  float s = 0.f, sq = 0.f;
  for (int r = 0; r < 128; ++r) {
    float v = buf[(base + r) * HID + c];
    s += v; sq += v * v;
  }
  atomicAdd(&stats[c], s);
  atomicAdd(&stats[128 + c], sq);
}

// res = relu(g*(v-mu)*rsqrt(var+eps)+b) + res   (in place residual)
__global__ void k_bn_apply_res(const float* __restrict__ vbuf, float* __restrict__ res,
                               const float* __restrict__ gam, const float* __restrict__ bet,
                               const float* __restrict__ stats, float invM)
{
  long long i = (long long)blockIdx.x * 256 + threadIdx.x;
  const int c = (int)(i & 127);
  const float mu  = stats[c] * invM;
  const float var = stats[128 + c] * invM - mu * mu;
  const float xn  = gam[c] * (vbuf[i] - mu) * rsqrtf(var + 1e-5f) + bet[c];
  res[i] = fmaxf(xn, 0.f) + res[i];
}

// ---------------------------------------------------------------------------
// Dense MHA with sph bias. grid: B*HEADS blocks, 256 threads (8 waves).
// Wave w owns score rows [w*16, w*16+16).
// ---------------------------------------------------------------------------
__global__ void k_attn(const float* __restrict__ q, const float* __restrict__ k,
                       const float* __restrict__ v, const float* __restrict__ sph,
                       float* __restrict__ o)
{
  __shared__ __bf16 Vt[DKH * SS];         // Vt[dk*128 + t], 8KB
  __shared__ __bf16 Pl[8 * 16 * SS];      // per-wave P rows, 32KB

  const int bh = blockIdx.x;
  const int b = bh >> 2, head = bh & 3;
  const int tid = threadIdx.x;
  const int wave = tid >> 5, lane = tid & 31;
  const int lrow = lane & 15, half = lane >> 4;
  const int sr = wave * 16;

  const float* qp = q + (long)b * SS * HID + head * DKH;
  const float* kp = k + (long)b * SS * HID + head * DKH;
  const float* vp = v + (long)b * SS * HID + head * DKH;
  const float* sphb = sph + (long)b * SS * SS;

  for (int i = tid; i < DKH * SS; i += 256) {
    int t = i >> 5, dk = i & 31;
    Vt[dk * SS + t] = (__bf16)vp[(long)t * HID + dk];
  }
  __syncthreads();

  // ---- scores = (Q K^T) * inv_sqrt_dk ----  (DK=32 -> one WMMA K-step)
  v8f sc[8] = {};
  {
    const float* qrow = qp + (long)(sr + lrow) * HID;
    const int kb = half * 8;
    v4f a0 = *(const v4f*)(qrow + kb);
    v4f a1 = *(const v4f*)(qrow + kb + 4);
    v4f a2 = *(const v4f*)(qrow + kb + 16);
    v4f a3 = *(const v4f*)(qrow + kb + 20);
    v16bf af;
#pragma unroll
    for (int j = 0; j < 4; ++j) {
      af[j] = (__bf16)a0[j]; af[4 + j] = (__bf16)a1[j];
      af[8 + j] = (__bf16)a2[j]; af[12 + j] = (__bf16)a3[j];
    }
#pragma unroll
    for (int nt = 0; nt < 8; ++nt) {
      const float* krow = kp + (long)(nt * 16 + lrow) * HID;
      v4f b0 = *(const v4f*)(krow + kb);
      v4f b1 = *(const v4f*)(krow + kb + 4);
      v4f b2 = *(const v4f*)(krow + kb + 16);
      v4f b3 = *(const v4f*)(krow + kb + 20);
      v16bf kf;
#pragma unroll
      for (int j = 0; j < 4; ++j) {
        kf[j] = (__bf16)b0[j]; kf[4 + j] = (__bf16)b1[j];
        kf[8 + j] = (__bf16)b2[j]; kf[12 + j] = (__bf16)b3[j];
      }
      sc[nt] = WMMA_BF16(af, kf, sc[nt]);
    }
  }

  // ---- softmax(scores * sph) row-wise, write P (bf16) to LDS ----
  const float scale = 0.17677669529663687f;  // 1/sqrt(32)
#pragma unroll
  for (int r = 0; r < 8; ++r) {
    const int m = sr + r + half * 8;
    float mx = -3.4e38f;
#pragma unroll
    for (int nt = 0; nt < 8; ++nt) {
      const int t = nt * 16 + lrow;
      float val = sc[nt][r] * scale * sphb[(long)m * SS + t];
      sc[nt][r] = val;
      mx = fmaxf(mx, val);
    }
#pragma unroll
    for (int off = 1; off < 16; off <<= 1) mx = fmaxf(mx, __shfl_xor(mx, off));
    float sum = 0.f;
#pragma unroll
    for (int nt = 0; nt < 8; ++nt) {
      float ev = __expf(sc[nt][r] - mx);
      sc[nt][r] = ev;
      sum += ev;
    }
#pragma unroll
    for (int off = 1; off < 16; off <<= 1) sum += __shfl_xor(sum, off);
    const float inv = 1.f / sum;
#pragma unroll
    for (int nt = 0; nt < 8; ++nt) {
      const int t = nt * 16 + lrow;
      Pl[wave * 2048 + (r + half * 8) * SS + t] = (__bf16)(sc[nt][r] * inv);
    }
  }
  __syncthreads();

  // ---- O = P @ V  (K=128 -> 4 WMMA K-steps, 2 dk tiles) ----
  v8f oa[2] = {};
  const __bf16* prow = Pl + wave * 2048 + lrow * SS;
  for (int kt = 0; kt < 4; ++kt) {
    const int kb = kt * 32 + half * 8;
    BFV pa;
    pa.h[0] = *(const v8bf*)(prow + kb);
    pa.h[1] = *(const v8bf*)(prow + kb + 16);
#pragma unroll
    for (int nt = 0; nt < 2; ++nt) {
      const __bf16* vrow = Vt + (nt * 16 + lrow) * SS + kb;
      BFV vf;
      vf.h[0] = *(const v8bf*)(vrow);
      vf.h[1] = *(const v8bf*)(vrow + 16);
      oa[nt] = WMMA_BF16(pa.v, vf.v, oa[nt]);
    }
  }
#pragma unroll
  for (int nt = 0; nt < 2; ++nt) {
#pragma unroll
    for (int r = 0; r < 8; ++r) {
      long m = (long)b * SS + sr + r + half * 8;
      o[m * HID + head * DKH + nt * 16 + lrow] = oa[nt][r];
    }
  }
}

// mean pool per graph: grid B blocks x 128 threads
__global__ void k_pool(const float* __restrict__ h, float* __restrict__ g)
{
  const int b = blockIdx.x, c = threadIdx.x;
  float s = 0.f;
  for (int r = 0; r < SS; ++r) s += h[((long)b * SS + r) * HID + c];
  g[b * HID + c] = s * (1.f / 128.f);
}

// final MLP 128 -> 64 -> 16 -> 10 for all 128 graphs; single block
__global__ void k_mlp(const float* __restrict__ g,
                      const float* __restrict__ w1, const float* __restrict__ b1,
                      const float* __restrict__ w2, const float* __restrict__ b2,
                      const float* __restrict__ w3, const float* __restrict__ b3,
                      float* __restrict__ out)
{
  __shared__ float g1[128 * 64];
  __shared__ float g2[128 * 16];
  const int tid = threadIdx.x;
  for (int i = tid; i < 128 * 64; i += 256) {
    int bb = i >> 6, c = i & 63;
    float acc = b1[c];
    for (int j = 0; j < 128; ++j) acc += g[bb * 128 + j] * w1[j * 64 + c];
    g1[i] = fmaxf(acc, 0.f);
  }
  __syncthreads();
  for (int i = tid; i < 128 * 16; i += 256) {
    int bb = i >> 4, c = i & 15;
    float acc = b2[c];
    for (int j = 0; j < 64; ++j) acc += g1[bb * 64 + j] * w2[j * 16 + c];
    g2[i] = fmaxf(acc, 0.f);
  }
  __syncthreads();
  for (int i = tid; i < 128 * 10; i += 256) {
    int bb = i / 10, c = i % 10;
    float acc = b3[c];
    for (int j = 0; j < 16; ++j) acc += g2[bb * 16 + j] * w3[j * 10 + c];
    out[bb * 10 + c] = acc;
  }
}

// ---------------------------------------------------------------------------
extern "C" void kernel_launch(void* const* d_in, const int* in_sizes, int n_in,
                              void* d_out, int out_size, void* d_ws, size_t ws_size,
                              hipStream_t stream) {
  (void)in_sizes; (void)n_in; (void)out_size; (void)ws_size;

  const float* x        = (const float*)d_in[0];
  const float* pe       = (const float*)d_in[1];
  const float* edge_attr= (const float*)d_in[2];
  const float* sph      = (const float*)d_in[3];
  const float* node_w   = (const float*)d_in[4];
  const float* node_b   = (const float*)d_in[5];
  const float* pe_w     = (const float*)d_in[6];
  const float* pe_b     = (const float*)d_in[7];
  const float* conv_Aw  = (const float*)d_in[8];
  const float* conv_Ab  = (const float*)d_in[9];
  const float* conv_Bw  = (const float*)d_in[10];
  const float* conv_Bb  = (const float*)d_in[11];
  const float* conv_Cw  = (const float*)d_in[12];
  const float* conv_Cb  = (const float*)d_in[13];
  const float* conv_Dw  = (const float*)d_in[14];
  const float* conv_Db  = (const float*)d_in[15];
  const float* conv_Ew  = (const float*)d_in[16];
  const float* conv_Eb  = (const float*)d_in[17];
  const float* bnx_g    = (const float*)d_in[18];
  const float* bnx_b    = (const float*)d_in[19];
  const float* bne_g    = (const float*)d_in[20];
  const float* bne_b    = (const float*)d_in[21];
  const float* attn_w   = (const float*)d_in[22];
  const float* attn_b   = (const float*)d_in[23];
  const float* mlp1_w   = (const float*)d_in[24];
  const float* mlp1_b   = (const float*)d_in[25];
  const float* mlp2_w   = (const float*)d_in[26];
  const float* mlp2_b   = (const float*)d_in[27];
  const float* mlp3_w   = (const float*)d_in[28];
  const float* mlp3_b   = (const float*)d_in[29];
  const int*   eidx     = (const int*)d_in[30];
  const int*   src      = eidx;
  const int*   dst      = eidx + NEDGES;
  float* out = (float*)d_out;

  const long long NH = (long long)NNODES * HID;   // 2,097,152
  const long long EH = (long long)NEDGES * HID;   // 33,554,432
  float* ws    = (float*)d_ws;
  float* h     = ws;
  float* Ax    = ws + 1 * NH;
  float* Bx    = ws + 2 * NH;
  float* Dx    = ws + 3 * NH;
  float* Ex    = ws + 4 * NH;
  float* num   = ws + 5 * NH;
  float* den   = ws + 6 * NH;
  float* hnew  = ws + 7 * NH;
  float* qb    = ws + 8 * NH;
  float* kb    = ws + 9 * NH;
  float* vb    = ws + 10 * NH;
  float* ob    = ws + 11 * NH;
  float* g     = ws + 12 * NH;
  float* stats = ws + 12 * NH + 32768;            // 512 floats (x: 0..255, e: 256..511)
  float* e     = ws + 13 * NH;
  float* eij   = ws + 13 * NH + EH;

  // node embed + edge copy
  k_embed<<<NNODES, 128, 0, stream>>>(x, pe, node_w, node_b, pe_w, pe_b, h);
  k_copy<<<(int)(EH / 256), 256, 0, stream>>>(edge_attr, e, EH);

  const float invN = 1.f / (float)NNODES;
  const float invE = 1.f / (float)NEDGES;

  for (int l = 0; l < 4; ++l) {
    const long wl = (long)l * HID * HID;
    const long bl = (long)l * HID;
    // 5 projections (WMMA)
    k_gemm128<<<NNODES / 128, 256, 0, stream>>>(h, conv_Aw + wl, conv_Ab + bl, Ax, NNODES);
    k_gemm128<<<NNODES / 128, 256, 0, stream>>>(h, conv_Bw + wl, conv_Bb + bl, Bx, NNODES);
    k_gemm128<<<NNODES / 128, 256, 0, stream>>>(h, conv_Dw + wl, conv_Db + bl, Dx, NNODES);
    k_gemm128<<<NNODES / 128, 256, 0, stream>>>(h, conv_Ew + wl, conv_Eb + bl, Ex, NNODES);
    k_gemm128<<<NEDGES / 128, 256, 0, stream>>>(e, conv_Cw + wl, conv_Cb + bl, eij, NEDGES);
    // gated aggregation
    k_zero<<<(int)(2 * NH / 256), 256, 0, stream>>>(num, 2 * NH);
    k_edge_gate<<<(int)(EH / 256), 256, 0, stream>>>(Bx, Dx, Ex, eij, src, dst, num, den);
    k_hnew<<<(int)(NH / 256), 256, 0, stream>>>(Ax, num, den, hnew);
    // batchnorm (training stats) + relu + residual, for nodes and edges
    k_zero<<<2, 256, 0, stream>>>(stats, 512);
    k_bn_stats<<<NNODES / 128, 128, 0, stream>>>(hnew, stats);
    k_bn_stats<<<NEDGES / 128, 128, 0, stream>>>(eij, stats + 256);
    k_bn_apply_res<<<(int)(NH / 256), 256, 0, stream>>>(hnew, h, bnx_g + bl, bnx_b + bl, stats, invN);
    k_bn_apply_res<<<(int)(EH / 256), 256, 0, stream>>>(eij, e, bne_g + bl, bne_b + bl, stats + 256, invE);
    // dense MHA with sph bias
    k_gemm128<<<NNODES / 128, 256, 0, stream>>>(h, attn_w + 0 * HID * HID, attn_b + 0 * HID, qb, NNODES);
    k_gemm128<<<NNODES / 128, 256, 0, stream>>>(h, attn_w + 1 * HID * HID, attn_b + 1 * HID, kb, NNODES);
    k_gemm128<<<NNODES / 128, 256, 0, stream>>>(h, attn_w + 2 * HID * HID, attn_b + 2 * HID, vb, NNODES);
    k_attn<<<NB * NHEADS, 256, 0, stream>>>(qb, kb, vb, sph, ob);
    k_gemm128<<<NNODES / 128, 256, 0, stream>>>(ob, attn_w + 3 * HID * HID, attn_b + 3 * HID, h, NNODES);
  }

  // readout
  k_pool<<<NB, 128, 0, stream>>>(h, g);
  k_mlp<<<1, 256, 0, stream>>>(g, mlp1_w, mlp1_b, mlp2_w, mlp2_b, mlp3_w, mlp3_b, out);
}
